// GraphTransformerInterpolantNet_34110630265294
// MI455X (gfx1250) — matrix operand order repaired and checked by reading, not compiled, not run on hardware
//
#include <hip/hip_runtime.h>
#include <cmath>

#define HEADS 8
#define EDIM 25

typedef __attribute__((ext_vector_type(16))) __bf16 v16bf;
typedef __attribute__((ext_vector_type(8)))  float  v8f;

// order-preserving float->uint encoding for atomicMax-based segment max
__device__ __forceinline__ unsigned encOrd(float v) {
  unsigned u = __builtin_bit_cast(unsigned, v);
  return (u & 0x80000000u) ? ~u : (u | 0x80000000u);
}
__device__ __forceinline__ float decOrd(unsigned e) {
  unsigned u = (e & 0x80000000u) ? (e ^ 0x80000000u) : ~e;
  return __builtin_bit_cast(float, u);
}

// ---------- WMMA GEMM: Y[M,Dout] = act(X[M,K] @ W[Dout,K]^T + bias) ----------
// 8 waves / block, each wave one 16x16 tile (same M-tile, 8 adjacent N-tiles).
// X tile (16x32 f32) staged once per block through LDS; A-fragment read as four
// aligned float4 runs per lane; B read as four float4s per lane. Native
// (__bf16) conversions feed v_wmma_f32_16x16x32_bf16. Only the K%32 tail
// (lin_in, K=16) takes the guarded slow path.
__global__ void __launch_bounds__(256)
k_gemm(const float* __restrict__ X, const float* __restrict__ W,
       const float* __restrict__ Bias, float* __restrict__ Y,
       int K, int Dout, int relu) {
  __shared__ __align__(16) float xs[16 * 32];
  const int tid   = threadIdx.x;
  const int lane  = tid & 31;
  const int wave  = tid >> 5;
  const int group = lane >> 4;      // 0: lanes 0-15, 1: lanes 16-31
  const int r     = lane & 15;
  const int m0    = blockIdx.x << 4;
  const int n0    = ((blockIdx.y << 3) + wave) << 4;
  const int ncol  = n0 + r;
  const bool colok = (ncol < Dout);
  const float* wrow = W + (size_t)(colok ? ncol : 0) * K;

  v8f acc = {0.f, 0.f, 0.f, 0.f, 0.f, 0.f, 0.f, 0.f};
  const int kfull = K & ~31;

  for (int k0 = 0; k0 < kfull; k0 += 32) {
    __syncthreads();
    {
      // 512 floats, 256 threads -> one float2 each (aligned: K,k0,cc all even)
      int i  = tid << 1;
      int rr = i >> 5, cc = i & 31;
      *reinterpret_cast<float2*>(xs + i) =
          *reinterpret_cast<const float2*>(X + (size_t)(m0 + rr) * K + k0 + cc);
    }
    __syncthreads();

    // A lane data: row r, K-offsets {8g..8g+7} and {16+8g..23+8g} (contiguous)
    const float* xr = xs + (r << 5) + (group << 3);
    float4 x0 = *reinterpret_cast<const float4*>(xr);
    float4 x1 = *reinterpret_cast<const float4*>(xr + 4);
    float4 x2 = *reinterpret_cast<const float4*>(xr + 16);
    float4 x3 = *reinterpret_cast<const float4*>(xr + 20);
    // B lane data: col ncol, K = k0 + {8g..8g+7} and {16+8g..23+8g} (16B aligned)
    const float* wp = wrow + k0 + (group << 3);
    float4 w0 = *reinterpret_cast<const float4*>(wp);
    float4 w1 = *reinterpret_cast<const float4*>(wp + 4);
    float4 w2 = *reinterpret_cast<const float4*>(wp + 16);
    float4 w3 = *reinterpret_cast<const float4*>(wp + 20);

    v16bf a, b;
    a[0]  = (__bf16)x0.x; a[1]  = (__bf16)x0.y; a[2]  = (__bf16)x0.z; a[3]  = (__bf16)x0.w;
    a[4]  = (__bf16)x1.x; a[5]  = (__bf16)x1.y; a[6]  = (__bf16)x1.z; a[7]  = (__bf16)x1.w;
    a[8]  = (__bf16)x2.x; a[9]  = (__bf16)x2.y; a[10] = (__bf16)x2.z; a[11] = (__bf16)x2.w;
    a[12] = (__bf16)x3.x; a[13] = (__bf16)x3.y; a[14] = (__bf16)x3.z; a[15] = (__bf16)x3.w;
    b[0]  = (__bf16)w0.x; b[1]  = (__bf16)w0.y; b[2]  = (__bf16)w0.z; b[3]  = (__bf16)w0.w;
    b[4]  = (__bf16)w1.x; b[5]  = (__bf16)w1.y; b[6]  = (__bf16)w1.z; b[7]  = (__bf16)w1.w;
    b[8]  = (__bf16)w2.x; b[9]  = (__bf16)w2.y; b[10] = (__bf16)w2.z; b[11] = (__bf16)w2.w;
    b[12] = (__bf16)w3.x; b[13] = (__bf16)w3.y; b[14] = (__bf16)w3.z; b[15] = (__bf16)w3.w;

    acc = __builtin_amdgcn_wmma_f32_16x16x32_bf16(false, a, false, b,
                                                  (short)0, acc, false, false);
  }

  if (kfull < K) {  // K%32 tail (only lin_in, K=16) -- uniform branch
    __syncthreads();
    for (int i = tid; i < 16 * 32; i += 256) {
      int rr = i >> 5, cc = i & 31;
      int kk = kfull + cc;
      xs[i] = (kk < K) ? X[(size_t)(m0 + rr) * K + kk] : 0.f;
    }
    __syncthreads();
    v16bf a, b;
#pragma unroll
    for (int j = 0; j < 16; ++j) {
      int p = j >> 1, lo = j & 1;
      int ka = ((p < 4) ? (p << 1) : (16 + ((p - 4) << 1))) + lo + (group << 3);
      a[j] = (__bf16)xs[(r << 5) + ka];
      int kb = kfull + ((j < 8) ? j : (8 + j)) + (group << 3);
      b[j] = (__bf16)((colok && kb < K) ? wrow[kb] : 0.f);
    }
    acc = __builtin_amdgcn_wmma_f32_16x16x32_bf16(false, a, false, b,
                                                  (short)0, acc, false, false);
  }

  if (colok) {
    float bias = Bias ? Bias[ncol] : 0.f;
#pragma unroll
    for (int vv = 0; vv < 8; ++vv) {
      int mm = m0 + vv + (group << 3);  // C/D: VGPR v -> M=v (+8 for lanes 16-31)
      float val = acc[vv] + bias;
      if (relu) val = fmaxf(val, 0.f);
      Y[(size_t)mm * Dout + ncol] = val;
    }
  }
}

// ---------- edge embedding: basis(16) + spherical harmonics(9) ----------
__global__ void k_edge_embed(const float* __restrict__ pos, const int* __restrict__ ei,
                             float* __restrict__ ea, int E) {
  int e = blockIdx.x * blockDim.x + threadIdx.x;
  if (e >= E) return;
  int s = ei[e], d = ei[E + e];
  float vx = pos[3 * s + 0] - pos[3 * d + 0];
  float vy = pos[3 * s + 1] - pos[3 * d + 1];
  float vz = pos[3 * s + 2] - pos[3 * d + 2];
  float len = sqrtf(vx * vx + vy * vy + vz * vz);
  const float step = 1.0f / 17.0f;                      // MAX_RADIUS/(N_BASIS+1)
  const float CB = 1.14136f * 7.38905609893065f * 4.0f; // 1.14136*e^2*sqrt(16)
  float* row = ea + (size_t)e * EDIM;
#pragma unroll
  for (int i = 0; i < 16; ++i) {
    float diff = (len - (float)(i + 1) * step) / step;
    float xp = diff + 1.0f, xm = 1.0f - diff;
    float a = (xp > 0.f) ? __expf(-1.0f / xp) : 0.f;
    float b = (xm > 0.f) ? __expf(-1.0f / xm) : 0.f;
    row[i] = CB * a * b;
  }
  float il = 1.0f / fmaxf(len, 1e-12f);
  float x = vx * il, y = vy * il, z = vz * il;
  const float s3 = 1.7320508075688772f, s5 = 2.2360679774997896f;
  row[16] = 1.f;
  row[17] = s3 * x;
  row[18] = s3 * y;
  row[19] = s3 * z;
  row[20] = s5 * s3 * x * z;
  row[21] = s5 * s3 * x * y;
  row[22] = s5 * (y * y - 0.5f * (x * x + z * z));
  row[23] = s5 * s3 * y * z;
  row[24] = s5 * (s3 * 0.5f) * (z * z - x * x);
}

__global__ void k_zero(float* __restrict__ p, size_t n) {
  size_t i = (size_t)blockIdx.x * blockDim.x + threadIdx.x;
  if (i < n) p[i] = 0.f;
}

// qproj[n,h,d] = sum_c q[n, h*C+c] * We[h*C+c, d]   (folds q . e analytically)
__global__ void k_qproj(const float* __restrict__ q, const float* __restrict__ We,
                        float* __restrict__ qp, int H, int C, size_t total) {
  size_t t = (size_t)blockIdx.x * blockDim.x + threadIdx.x;
  if (t >= total) return;
  int d  = (int)(t % EDIM);
  size_t nh = t / EDIM;
  int h  = (int)(nh % HEADS);
  size_t n = nh / HEADS;
  const float* qr = q + n * H + h * C;
  const float* wr = We + (size_t)(h * C) * EDIM + d;
  float acc = 0.f;
  for (int c = 0; c < C; ++c) acc += qr[c] * wr[(size_t)c * EDIM];
  qp[t] = acc;
}

// alpha logit per (edge, head) + running segment max over dst
__global__ void k_alpha(const float* __restrict__ q, const float* __restrict__ k,
                        const float* __restrict__ qp, const float* __restrict__ ea,
                        const int* __restrict__ ei, float* __restrict__ alpha,
                        unsigned* __restrict__ amax, int E, int H, int C, float scale) {
  size_t t = (size_t)blockIdx.x * blockDim.x + threadIdx.x;
  if (t >= (size_t)E * HEADS) return;
  int h = (int)(t % HEADS);
  int e = (int)(t / HEADS);
  int s = ei[e], d = ei[E + e];
  const float* qr = q + (size_t)d * H + h * C;
  const float* kr = k + (size_t)s * H + h * C;
  float acc = 0.f;
  for (int c = 0; c < C; ++c) acc += qr[c] * kr[c];
  const float* qpr = qp + ((size_t)d * HEADS + h) * EDIM;
  const float* er  = ea + (size_t)e * EDIM;
  for (int i = 0; i < EDIM; ++i) acc += qpr[i] * er[i];
  acc *= scale;
  alpha[t] = acc;
  atomicMax(&amax[d * HEADS + h], encOrd(acc));
}

// exp(alpha - max) and segment-sum denominator
__global__ void k_expnorm(const int* __restrict__ ei, float* __restrict__ alpha,
                          const unsigned* __restrict__ amax, float* __restrict__ denom,
                          int E) {
  size_t t = (size_t)blockIdx.x * blockDim.x + threadIdx.x;
  if (t >= (size_t)E * HEADS) return;
  int h = (int)(t % HEADS);
  int e = (int)(t / HEADS);
  int d = ei[E + e];
  float ex = __expf(alpha[t] - decOrd(amax[d * HEADS + h]));
  alpha[t] = ex;
  atomicAdd(&denom[d * HEADS + h], ex);
}

// out[dst] += w * v[src]  and  S[dst,h,:] += w * ea[e,:]
__global__ void k_msg(const int* __restrict__ ei, const float* __restrict__ alpha,
                      const float* __restrict__ denom, const float* __restrict__ v,
                      const float* __restrict__ ea, float* __restrict__ out,
                      float* __restrict__ S, int E, int H, int C) {
  size_t t = (size_t)blockIdx.x * blockDim.x + threadIdx.x;
  if (t >= (size_t)E * HEADS) return;
  int h = (int)(t % HEADS);
  int e = (int)(t / HEADS);
  int s = ei[e], d = ei[E + e];
  float w = alpha[t] / denom[d * HEADS + h];
  const float* vr = v + (size_t)s * H + h * C;
  float* orow = out + (size_t)d * H + h * C;
  for (int c = 0; c < C; ++c) atomicAdd(&orow[c], w * vr[c]);
  const float* er = ea + (size_t)e * EDIM;
  float* Sr = S + ((size_t)d * HEADS + h) * EDIM;
  for (int i = 0; i < EDIM; ++i) atomicAdd(&Sr[i], w * er[i]);
}

// out[n,col] += S[n,h,:] . We[col,:]  then ReLU  (reconstructs alpha*e messages)
__global__ void k_ecorrect(float* __restrict__ out, const float* __restrict__ S,
                           const float* __restrict__ We, int H, int C, int relu,
                           size_t total) {
  size_t t = (size_t)blockIdx.x * blockDim.x + threadIdx.x;
  if (t >= total) return;
  int col = (int)(t % H);
  size_t n = t / H;
  int h = col / C;
  const float* Sr = S + (n * HEADS + h) * EDIM;
  const float* wr = We + (size_t)col * EDIM;
  float acc = out[t];
  for (int i = 0; i < EDIM; ++i) acc += Sr[i] * wr[i];
  out[t] = relu ? fmaxf(acc, 0.f) : acc;
}

// ---------- host orchestration ----------
struct Lin  { const float* W; const float* b; };
struct Conv { Lin q, k, v; const float* eW; Lin skip; };
struct Net  { Conv conv[3]; Lin mlp1, mlp2; };

static inline void gemm(hipStream_t st, const float* X, const Lin& L, float* Y,
                        int M, int K, int Dout, int relu) {
  dim3 grid(M / 16, (((Dout + 15) / 16) + 7) / 8);
  k_gemm<<<grid, 256, 0, st>>>(X, L.W, L.b, Y, K, Dout, relu);
}

static void run_conv(hipStream_t st, const float* x, float* out, const Conv& cv,
                     int N, int E, int din, int H, const int* ei, const float* ea,
                     float* qb, float* kb, float* vb, float* alpha,
                     unsigned* amax, float* denom, float* S, float* qp) {
  const int C = H / HEADS;
  gemm(st, x, cv.q, qb, N, din, H, 0);
  gemm(st, x, cv.k, kb, N, din, H, 0);
  gemm(st, x, cv.v, vb, N, din, H, 0);
  gemm(st, x, cv.skip, out, N, din, H, 0);      // skip connection seeds out
  // amax | denom | S are contiguous in workspace -> single zero fill
  size_t zc = (size_t)N * HEADS * (2 + EDIM);
  k_zero<<<(unsigned)((zc + 255) / 256), 256, 0, st>>>((float*)amax, zc);
  size_t qpc = (size_t)N * HEADS * EDIM;
  k_qproj<<<(unsigned)((qpc + 255) / 256), 256, 0, st>>>(qb, cv.eW, qp, H, C, qpc);
  size_t ec = (size_t)E * HEADS;
  float scale = 1.0f / sqrtf((float)C);
  k_alpha<<<(unsigned)((ec + 255) / 256), 256, 0, st>>>(qb, kb, qp, ea, ei, alpha,
                                                        amax, E, H, C, scale);
  k_expnorm<<<(unsigned)((ec + 255) / 256), 256, 0, st>>>(ei, alpha, amax, denom, E);
  k_msg<<<(unsigned)((ec + 255) / 256), 256, 0, st>>>(ei, alpha, denom, vb, ea, out,
                                                      S, E, H, C);
  size_t nc = (size_t)N * H;
  k_ecorrect<<<(unsigned)((nc + 255) / 256), 256, 0, st>>>(out, S, cv.eW, H, C, 1, nc);
}

static void run_net(hipStream_t st, const float* x0, const Net& net, int din0, int H,
                    float* outF, int doutF, int N, int E, const int* ei,
                    const float* ea, float* bufA, float* bufB, float* qb, float* kb,
                    float* vb, float* alpha, unsigned* amax, float* denom, float* S,
                    float* qp) {
  const float* cur = x0;
  for (int l = 0; l < 3; ++l) {
    float* dst = (cur == bufA) ? bufB : bufA;
    run_conv(st, cur, dst, net.conv[l], N, E, (l == 0 ? din0 : H), H, ei, ea,
             qb, kb, vb, alpha, amax, denom, S, qp);
    cur = dst;
  }
  gemm(st, cur, net.mlp1, qb, N, H, H, 1);       // relu(mlp1)
  gemm(st, qb, net.mlp2, outF, N, H, doutF, 0);  // mlp2
}

extern "C" void kernel_launch(void* const* d_in, const int* in_sizes, int n_in,
                              void* d_out, int out_size, void* d_ws, size_t ws_size,
                              hipStream_t stream) {
  (void)n_in; (void)out_size; (void)ws_size;
  const int N = in_sizes[0] / 16;   // 8192 nodes (IN_DIM=16)
  const int E = in_sizes[3] / 2;    // 196608 edges
  const float* f   = (const float*)d_in[0];
  const float* pos = (const float*)d_in[1];
  const int*   ei  = (const int*)d_in[3];

  // params in setup_inputs() dict insertion order, depth-first
  int idx = 4;
  auto rl = [&](bool bias) {
    Lin l; l.W = (const float*)d_in[idx++];
    l.b = bias ? (const float*)d_in[idx++] : nullptr;
    return l;
  };
  Lin lin_in = rl(true);
  auto rnet = [&]() {
    Net n;
    for (int i = 0; i < 3; ++i) {
      n.conv[i].q = rl(true); n.conv[i].k = rl(true); n.conv[i].v = rl(true);
      n.conv[i].eW = (const float*)d_in[idx++];  // e has no bias
      n.conv[i].skip = rl(true);
    }
    n.mlp1 = rl(true); n.mlp2 = rl(true);
    return n;
  };
  Net cond = rnet(), netb = rnet(), neteta = rnet();

  // workspace carve-up (floats)
  float* ws = (float*)d_ws;
  size_t off = 0;
  auto alloc = [&](size_t cnt) { float* p = ws + off; off += cnt; return p; };
  float* ea    = alloc((size_t)E * EDIM);          // 19.7 MB
  float* bufA  = alloc((size_t)N * 512);
  float* bufB  = alloc((size_t)N * 512);
  float* qb    = alloc((size_t)N * 512);
  float* kb    = alloc((size_t)N * 512);
  float* vb    = alloc((size_t)N * 512);
  float* alpha = alloc((size_t)E * HEADS);
  float* amaxf = alloc((size_t)N * HEADS);         // encoded-uint segment max
  float* denom = alloc((size_t)N * HEADS);         // contiguous with amax
  float* S     = alloc((size_t)N * HEADS * EDIM);  // contiguous with denom
  float* qp    = alloc((size_t)N * HEADS * EDIM);

  k_edge_embed<<<(E + 255) / 256, 256, 0, stream>>>(pos, ei, ea, E);
  gemm(stream, f, lin_in, bufA, N, 16, 128, 0);    // h = f @ lin_in

  float* f_cond  = (float*)d_out;                  // N x 256
  float* b_out   = f_cond + (size_t)N * 256;       // N x 3
  float* eta_out = b_out + (size_t)N * 3;          // N x 3

  run_net(stream, bufA, cond, 128, 512, f_cond, 256, N, E, ei, ea,
          bufA, bufB, qb, kb, vb, alpha, (unsigned*)amaxf, denom, S, qp);
  run_net(stream, f_cond, netb, 256, 256, b_out, 3, N, E, ei, ea,
          bufA, bufB, qb, kb, vb, alpha, (unsigned*)amaxf, denom, S, qp);
  run_net(stream, f_cond, neteta, 256, 256, eta_out, 3, N, E, ei, ea,
          bufA, bufB, qb, kb, vb, alpha, (unsigned*)amaxf, denom, S, qp);
}
